// Extractor_5755256176962
// MI455X (gfx1250) — matrix operand order repaired and checked by reading, not compiled, not run on hardware
//
#include <hip/hip_runtime.h>
#include <hip/hip_bf16.h>

// ---------------------------------------------------------------------------
// CDNA5 (gfx1250, wave32) implementation of the cross-attention resblock.
// All GEMMs + attention use v_wmma_f32_16x16x32_f16 (f16 in, f32 accumulate).
// ---------------------------------------------------------------------------

typedef __attribute__((ext_vector_type(16))) _Float16 v16h;
typedef __attribute__((ext_vector_type(4)))  _Float16 v4h;
typedef __attribute__((ext_vector_type(8)))  float    v8f;

union FragH { v16h v; uint4 q[2]; _Float16 h[16]; };

__device__ __forceinline__ v8f wmma_f16(const FragH& a, const FragH& b, v8f c) {
  // 8 args: (neg_a, A, neg_b, B, c_mod, C, reuse_a, reuse_b)
  return __builtin_amdgcn_wmma_f32_16x16x32_f16(false, a.v, false, b.v,
                                                (short)0, c, false, false);
}

// ---------------------------------------------------------------------------
// fp32 -> fp16 conversion (weights), vectorized x4
// ---------------------------------------------------------------------------
__global__ __launch_bounds__(256)
void k_cvt_f16(const float* __restrict__ s, _Float16* __restrict__ d, int n4) {
  int i = blockIdx.x * 256 + threadIdx.x;
  if (i < n4) {
    float4 f = ((const float4*)s)[i];
    v4h h;
    h.x = (_Float16)f.x; h.y = (_Float16)f.y;
    h.z = (_Float16)f.z; h.w = (_Float16)f.w;
    ((v4h*)d)[i] = h;
  }
}

// ---------------------------------------------------------------------------
// LayerNorm over last dim, output f16.  One block (256 thr) per row.
// ---------------------------------------------------------------------------
__global__ __launch_bounds__(256)
void k_ln_f16(const float* __restrict__ x, const float* __restrict__ g,
              const float* __restrict__ b, _Float16* __restrict__ out, int C) {
  __shared__ float red0[256], red1[256];
  const int row = blockIdx.x, tid = threadIdx.x;
  const float* xr = x + (size_t)row * C;
  float s = 0.f, s2 = 0.f;
  for (int i = tid; i < C; i += 256) { float v = xr[i]; s += v; s2 += v * v; }
  red0[tid] = s; red1[tid] = s2;
  __syncthreads();
  for (int off = 128; off > 0; off >>= 1) {
    if (tid < off) { red0[tid] += red0[tid + off]; red1[tid] += red1[tid + off]; }
    __syncthreads();
  }
  const float mean = red0[0] / (float)C;
  const float var  = red1[0] / (float)C - mean * mean;
  const float rstd = rsqrtf(var + 1e-5f);
  _Float16* orow = out + (size_t)row * C;
  for (int i = tid; i < C; i += 256)
    orow[i] = (_Float16)((xr[i] - mean) * rstd * g[i] + b[i]);
}

// ---------------------------------------------------------------------------
// Tiled WMMA GEMM:  out[m][n] = sum_k A[m][k] * W[n][k]  (+bias +gelu +resid)
// A: [M,K] f16 row-major;  W: [N,K] f16 row-major (i.e. the X @ W^T form).
// Block = 128 threads (4 waves); macro tile 64x64; each wave does 32x32.
// flags bit0: QuickGELU before residual/store.
// ---------------------------------------------------------------------------
#define BK 32
__global__ __launch_bounds__(128)
void k_gemm(const _Float16* __restrict__ A, const _Float16* __restrict__ W,
            const float* __restrict__ bias, const float* __restrict__ resid,
            float* __restrict__ outF, _Float16* __restrict__ outH,
            int M, int N, int K, int flags) {
  __shared__ _Float16 As[64][BK + 8];   // row stride 80 B (16B aligned)
  __shared__ _Float16 Ws[64][BK + 8];
  const int tid  = threadIdx.x;
  const int lane = tid & 31, wave = tid >> 5;
  const int bm = blockIdx.y * 64, bn = blockIdx.x * 64;
  const int wm = (wave >> 1) * 32, wn = (wave & 1) * 32;
  const int nn = lane & 15, kg = lane >> 4;     // N-lane / K-group of wave32
  v8f acc[2][2] = {};

  const int lr = tid >> 1;            // 0..63 : tile row loaded by this thread
  const int lc = (tid & 1) * 16;      // 0/16  : K-chunk
  const _Float16* Ag = A + (size_t)(bm + lr) * K + lc;
  const _Float16* Wg = W + (size_t)(bn + lr) * K + lc;

  for (int k0 = 0; k0 < K; k0 += BK) {
    uint4 a0 = *(const uint4*)(Ag);
    uint4 a1 = *(const uint4*)(Ag + 8);
    uint4 w0 = *(const uint4*)(Wg);
    uint4 w1 = *(const uint4*)(Wg + 8);
    __builtin_prefetch(Ag + BK, 0, 3);
    __builtin_prefetch(Wg + BK, 0, 3);
    __syncthreads();
    *(uint4*)&As[lr][lc]     = a0;  *(uint4*)&As[lr][lc + 8] = a1;
    *(uint4*)&Ws[lr][lc]     = w0;  *(uint4*)&Ws[lr][lc + 8] = w1;
    __syncthreads();
    Ag += BK; Wg += BK;

    FragH a[2], b[2];
#pragma unroll
    for (int i = 0; i < 2; i++) {
      // A 16x32 frag: lane kg half picks K {kg*8..+7, 16+kg*8..+7}
      const _Float16* pa = &As[wm + i * 16 + nn][kg * 8];
      a[i].q[0] = *(const uint4*)pa;
      a[i].q[1] = *(const uint4*)(pa + 16);
      // B 32x16 frag: lane nn = column (W row), K {kg*8..+7, 16+kg*8..+7}
      const _Float16* pb = &Ws[wn + i * 16 + nn][kg * 8];
      b[i].q[0] = *(const uint4*)pb;
      b[i].q[1] = *(const uint4*)(pb + 16);
    }
#pragma unroll
    for (int i = 0; i < 2; i++)
#pragma unroll
      for (int j = 0; j < 2; j++)
        acc[i][j] = wmma_f16(a[i], b[j], acc[i][j]);
  }

  // Epilogue.  C layout: lanes 0-15 VGPR v -> (M=v, N=lane); lanes 16-31 -> M=8+v.
  const int mh = kg;
#pragma unroll
  for (int i = 0; i < 2; i++)
#pragma unroll
    for (int j = 0; j < 2; j++) {
      const int col = bn + wn + j * 16 + nn;
      const float bv = bias ? bias[col] : 0.f;
#pragma unroll
      for (int v = 0; v < 8; v++) {
        const int row = bm + wm + i * 16 + v + 8 * mh;
        float val = acc[i][j][v] + bv;
        if (flags & 1) val = val / (1.f + __expf(-1.702f * val)); // QuickGELU
        if (resid) val += resid[(size_t)row * N + col];
        if (outF) outF[(size_t)row * N + col] = val;
        if (outH) outH[(size_t)row * N + col] = (_Float16)val;
      }
    }
}

// ---------------------------------------------------------------------------
// Flash attention, one (n,h) head-slice per blockIdx.y, 64 query rows per
// block (16 per wave).  q/k/v layout: [(t*NB + n)*1024 + h*64 + d], f16.
// S = (Q K^T)/8, online softmax over Ty=2048 in chunks of 64, O = P V.
// ---------------------------------------------------------------------------
#define TS 64
__global__ __launch_bounds__(128)
void k_attn(const _Float16* __restrict__ q, const _Float16* __restrict__ k,
            const _Float16* __restrict__ v, _Float16* __restrict__ o) {
  __shared__ _Float16 Ks[TS][64 + 8];      // K chunk, [s][d], stride 144 B
  __shared__ _Float16 Vt[64][TS + 8];      // V chunk transposed, [d][s]
  __shared__ _Float16 Ps[4][16][TS + 8];   // per-wave P staging (A-layout feed)
  const int tid  = threadIdx.x;
  const int lane = tid & 31, wave = tid >> 5;
  const int nn = lane & 15, kg = lane >> 4, mh = kg;
  const int nh = blockIdx.y;              // 0..127
  const int n = nh >> 4, h = nh & 15;
  const int qrow0 = blockIdx.x * 64 + wave * 16;
  const size_t rs = 8 * 1024;             // t-stride in halfs
  const _Float16* qb = q + (size_t)n * 1024 + h * 64;
  const _Float16* kb = k + (size_t)n * 1024 + h * 64;
  const _Float16* vb = v + (size_t)n * 1024 + h * 64;

  // Q fragments (16 rows x 64 d = 2 K-steps), loaded once from global.
  FragH qa[2];
  {
    const _Float16* p = qb + (size_t)(qrow0 + nn) * rs;
    qa[0].q[0] = *(const uint4*)(p + kg * 8);
    qa[0].q[1] = *(const uint4*)(p + 16 + kg * 8);
    qa[1].q[0] = *(const uint4*)(p + 32 + kg * 8);
    qa[1].q[1] = *(const uint4*)(p + 48 + kg * 8);
  }

  v8f Oacc[4] = {};
  float mrow[8], lrow[8];
#pragma unroll
  for (int r = 0; r < 8; r++) { mrow[r] = -3.0e38f; lrow[r] = 0.f; }
  const float scale = 0.125f;             // 1/sqrt(64)

  for (int s0 = 0; s0 < 2048; s0 += TS) {
    __syncthreads();
    {   // cooperative chunk load: K 64 rows x d {0-15, 32-47}, V(T) 64x64 halves
      const int r = tid >> 1, c = (tid & 1) * 32;
      const _Float16* pk = kb + (size_t)(s0 + r) * rs + c;
      *(uint4*)&Ks[r][c]     = *(const uint4*)(pk);
      *(uint4*)&Ks[r][c + 8] = *(const uint4*)(pk + 8);
      const _Float16* pv = vb + (size_t)(s0 + r) * rs + c;
#pragma unroll
      for (int j = 0; j < 32; j++) Vt[c + j][r] = pv[j];
    }
    __syncthreads();

    // ---- S = Q K^T (4 n-tiles of 16 s-cols, 2 K-steps over d) ----
    v8f S[4];
#pragma unroll
    for (int j = 0; j < 4; j++) {
      v8f z = {};
      FragH b0, b1;
      const _Float16* pb = &Ks[j * 16 + nn][kg * 8];
      b0.q[0] = *(const uint4*)pb;  b0.q[1] = *(const uint4*)(pb + 16);
      pb = &Ks[j * 16 + nn][32 + kg * 8];
      b1.q[0] = *(const uint4*)pb;  b1.q[1] = *(const uint4*)(pb + 16);
      z = wmma_f16(qa[0], b0, z);
      z = wmma_f16(qa[1], b1, z);
      S[j] = z;
    }
#pragma unroll
    for (int j = 0; j < 4; j++)
#pragma unroll
      for (int r = 0; r < 8; r++) S[j][r] = S[j][r] * scale;

    // ---- online softmax: row max over 64 cols (4 tiles x 16 lanes) ----
#pragma unroll
    for (int r = 0; r < 8; r++) {
      float t = fmaxf(fmaxf(S[0][r], S[1][r]), fmaxf(S[2][r], S[3][r]));
#pragma unroll
      for (int msk = 1; msk < 16; msk <<= 1) t = fmaxf(t, __shfl_xor(t, msk, 32));
      const float mnew = fmaxf(mrow[r], t);
      const float alpha = __expf(mrow[r] - mnew);
      lrow[r] *= alpha;
#pragma unroll
      for (int jd = 0; jd < 4; jd++) Oacc[jd][r] = Oacc[jd][r] * alpha;
      mrow[r] = mnew;
    }
    // ---- P = exp(S-m): stage to LDS in A-layout order, accumulate l ----
#pragma unroll
    for (int r = 0; r < 8; r++) {
      float acc_s = 0.f;
#pragma unroll
      for (int j = 0; j < 4; j++) {
        const float p = __expf(S[j][r] - mrow[r]);
        acc_s += p;
        Ps[wave][r + 8 * mh][j * 16 + nn] = (_Float16)p;
      }
#pragma unroll
      for (int msk = 1; msk < 16; msk <<= 1) acc_s += __shfl_xor(acc_s, msk, 32);
      lrow[r] += acc_s;
    }
    // ---- O += P V  (A = P 16x64, B = V chunk via Vt) ----
    FragH pa[2];
#pragma unroll
    for (int ss = 0; ss < 2; ss++) {
      const _Float16* pp = &Ps[wave][nn][ss * 32 + kg * 8];
      pa[ss].q[0] = *(const uint4*)pp;
      pa[ss].q[1] = *(const uint4*)(pp + 16);
    }
#pragma unroll
    for (int jd = 0; jd < 4; jd++) {
      FragH b0, b1;
      const _Float16* pb = &Vt[jd * 16 + nn][kg * 8];
      b0.q[0] = *(const uint4*)pb;  b0.q[1] = *(const uint4*)(pb + 16);
      pb = &Vt[jd * 16 + nn][32 + kg * 8];
      b1.q[0] = *(const uint4*)pb;  b1.q[1] = *(const uint4*)(pb + 16);
      Oacc[jd] = wmma_f16(pa[0], b0, Oacc[jd]);
      Oacc[jd] = wmma_f16(pa[1], b1, Oacc[jd]);
    }
  }

  // ---- write O / l ----
  _Float16* ob = o + (size_t)n * 1024 + h * 64;
#pragma unroll
  for (int jd = 0; jd < 4; jd++)
#pragma unroll
    for (int r = 0; r < 8; r++) {
      const int t = qrow0 + r + 8 * mh;
      const int d = jd * 16 + nn;
      ob[(size_t)t * rs + d] = (_Float16)(Oacc[jd][r] / lrow[r]);
    }
}

// ---------------------------------------------------------------------------
// Host-side orchestration
// ---------------------------------------------------------------------------
extern "C" void kernel_launch(void* const* d_in, const int* in_sizes, int n_in,
                              void* d_out, int out_size, void* d_ws, size_t ws_size,
                              hipStream_t stream) {
  (void)in_sizes; (void)n_in; (void)out_size; (void)ws_size;
  const float* x         = (const float*)d_in[0];
  const float* y         = (const float*)d_in[1];
  const float* in_proj_w = (const float*)d_in[2];
  const float* in_proj_b = (const float*)d_in[3];
  const float* out_w     = (const float*)d_in[4];
  const float* out_b     = (const float*)d_in[5];
  const float* ln0_g     = (const float*)d_in[6];
  const float* ln0_b     = (const float*)d_in[7];
  const float* red_w     = (const float*)d_in[8];
  const float* red_b     = (const float*)d_in[9];
  const float* ln1_g     = (const float*)d_in[10];
  const float* ln1_b     = (const float*)d_in[11];
  const float* ln2_g     = (const float*)d_in[12];
  const float* ln2_b     = (const float*)d_in[13];
  const float* ln3_g     = (const float*)d_in[14];
  const float* ln3_b     = (const float*)d_in[15];
  const float* fc_w      = (const float*)d_in[16];
  const float* fc_b      = (const float*)d_in[17];
  const float* proj_w    = (const float*)d_in[18];
  const float* proj_b    = (const float*)d_in[19];
  float* outp = (float*)d_out;

  constexpr int D = 1024, QDIM = 1408, TX = 256, NB = 8, DMLP = 4096;
  const int Mx = TX * NB;        // 2048 query tokens*batch
  const int My = 2048 * NB;      // 16384 kv tokens*batch

  // ---- workspace bump allocator (~156 MB with reuse) ----
  char* wsp = (char*)d_ws;
  auto alloc = [&](size_t bytes) -> char* {
    char* p = wsp; wsp += (bytes + 255) & ~(size_t)255; return p;
  };
  _Float16* redw_h = (_Float16*)alloc((size_t)D * QDIM * 2);
  _Float16* inpw_h = (_Float16*)alloc((size_t)3 * D * D * 2);
  _Float16* outw_h = (_Float16*)alloc((size_t)D * D * 2);
  _Float16* fcw_h  = (_Float16*)alloc((size_t)DMLP * D * 2);
  _Float16* pjw_h  = (_Float16*)alloc((size_t)D * DMLP * 2);
  _Float16* xn_h   = (_Float16*)alloc((size_t)Mx * QDIM * 2);  // reused as q
  float*    x_red  = (float*)   alloc((size_t)Mx * D * 4);
  _Float16* xq_h   = (_Float16*)alloc((size_t)Mx * D * 2);     // reused: o, x2n
  _Float16* yk_h   = (_Float16*)alloc((size_t)My * D * 2);     // reused as h
  _Float16* k_h    = (_Float16*)alloc((size_t)My * D * 2);
  _Float16* v_h    = (_Float16*)alloc((size_t)My * D * 2);
  float*    x2     = (float*)   alloc((size_t)Mx * D * 4);
  _Float16* q_h   = xn_h;       // xn dead after reduction GEMM
  _Float16* o_h   = xq_h;       // xq dead after q GEMM
  _Float16* x2n_h = xq_h;       // o dead after out-proj GEMM
  _Float16* h_h   = yk_h;       // yk dead after k/v GEMMs (16.8MB <= 33.6MB)

  auto cvt = [&](const float* s, _Float16* d, size_t n) {
    int n4 = (int)(n / 4);
    k_cvt_f16<<<(n4 + 255) / 256, 256, 0, stream>>>(s, d, n4);
  };
  auto gemm = [&](const _Float16* A, const _Float16* W, const float* bias,
                  const float* resid, float* oF, _Float16* oH,
                  int M, int N, int K, int flags) {
    dim3 g(N / 64, M / 64);
    k_gemm<<<g, 128, 0, stream>>>(A, W, bias, resid, oF, oH, M, N, K, flags);
  };

  // 1) weights -> f16 (cheap vs. compute; keeps launch deterministic)
  cvt(red_w,     redw_h, (size_t)D * QDIM);
  cvt(in_proj_w, inpw_h, (size_t)3 * D * D);
  cvt(out_w,     outw_h, (size_t)D * D);
  cvt(fc_w,      fcw_h,  (size_t)DMLP * D);
  cvt(proj_w,    pjw_h,  (size_t)D * DMLP);

  // 2) x = LN0(x) @ red_w^T + red_b   -> x_red (fp32 residual base)
  k_ln_f16<<<Mx, 256, 0, stream>>>(x, ln0_g, ln0_b, xn_h, QDIM);
  gemm(xn_h, redw_h, red_b, nullptr, x_red, nullptr, Mx, D, QDIM, 0);

  // 3) LN2(x_red) -> xq ; LN3(y) -> yk
  k_ln_f16<<<Mx, 256, 0, stream>>>(x_red, ln2_g, ln2_b, xq_h, D);
  k_ln_f16<<<My, 256, 0, stream>>>(y,     ln3_g, ln3_b, yk_h, D);

  // 4) q/k/v projections
  gemm(xq_h, inpw_h,             in_proj_b,         nullptr, nullptr, q_h, Mx, D, D, 0);
  gemm(yk_h, inpw_h + (size_t)D * D,     in_proj_b + D,     nullptr, nullptr, k_h, My, D, D, 0);
  gemm(yk_h, inpw_h + (size_t)2 * D * D, in_proj_b + 2 * D, nullptr, nullptr, v_h, My, D, D, 0);

  // 5) flash attention over 128 (n,h) head-slices, 4 query tiles of 64
  k_attn<<<dim3(TX / 64, NB * 16), 128, 0, stream>>>(q_h, k_h, v_h, o_h);

  // 6) x2 = x_red + o @ out_w^T + out_b
  gemm(o_h, outw_h, out_b, x_red, x2, nullptr, Mx, D, D, 0);

  // 7) MLP: h = quick_gelu(LN1(x2) @ fc_w^T + fc_b); out = x2 + h @ proj_w^T + proj_b
  k_ln_f16<<<Mx, 256, 0, stream>>>(x2, ln1_g, ln1_b, x2n_h, D);
  gemm(x2n_h, fcw_h, fc_b,   nullptr, nullptr, h_h, Mx, DMLP, D, 1);
  gemm(h_h,   pjw_h, proj_b, x2,      outp,    nullptr, Mx, D, DMLP, 0);
}
